// NAT_40707700032264
// MI455X (gfx1250) — matrix operand order repaired and checked by reading, not compiled, not run on hardware
//
#include <hip/hip_runtime.h>
#include <stdint.h>

typedef __bf16 bf16;
typedef bf16 v16bf __attribute__((ext_vector_type(16)));
typedef bf16 v8bf  __attribute__((ext_vector_type(8)));
typedef float v8f  __attribute__((ext_vector_type(8)));

#define BN 2
#define CH 128
#define HH 96
#define WW 96
#define HEADS 4
#define HD 32
#define MROWS (BN*HH*WW)   // 18432, multiple of 32

static __device__ __forceinline__ bf16 f2bf(float f) {
    union { float f; uint32_t u; } in; in.f = f;
    uint32_t u = in.u;
    uint32_t r = (u + 0x7FFFu + ((u >> 16) & 1u)) >> 16;
    union { unsigned short s; bf16 b; } out; out.s = (unsigned short)r;
    return out.b;
}

// x [B,C,1,H,W] fp32 -> Xbf [B*H*W, C] bf16
__global__ void nat_cvt_x(const float* __restrict__ x, bf16* __restrict__ xb) {
    int i = blockIdx.x * blockDim.x + threadIdx.x;
    if (i >= BN * CH * HH * WW) return;
    int w = i % WW;
    int h = (i / WW) % HH;
    int d = (i / (WW * HH)) % CH;
    int b = i / (WW * HH * CH);
    xb[((b * HH + h) * WW + w) * CH + d] = f2bf(x[i]);
}

__global__ void nat_cvt(const float* __restrict__ s, bf16* __restrict__ d, int n) {
    int i = blockIdx.x * blockDim.x + threadIdx.x;
    if (i < n) d[i] = f2bf(s[i]);
}

// Y[M,N] = A[M,128](bf16) * W[N,128](bf16)^T + bias[N]
// Each wave computes a 32x64 output tile: 2 M-subtiles x 4 N-subtiles,
// 8 f32 accumulators, 8 WMMAs per 32-deep k-step (4 k-steps, K=128).
// MODE 0: fp32 out[M,N]; MODE 1: bf16 out[M,N];
// MODE 2: fp32 out in [B,C,H,W] layout with residual add.
template<int NCOLS, int MODE>
__global__ __launch_bounds__(256) void nat_gemm(
    const bf16* __restrict__ A, const bf16* __restrict__ W,
    const float* __restrict__ bias,
    float* __restrict__ outF, bf16* __restrict__ outB,
    const float* __restrict__ resid)
{
    const int tilesN = NCOLS / 64;                    // N supertiles (64 wide)
    const int totalTiles = (MROWS / 32) * tilesN;     // M supertiles (32 tall)
    int wave = blockIdx.x * (blockDim.x >> 5) + (threadIdx.x >> 5);
    if (wave >= totalTiles) return;
    int lane = threadIdx.x & 31;
    int tm = wave / tilesN, tn = wave % tilesN;
    int hi = lane >> 4;      // half-wave select
    int ln = lane & 15;

    const bf16* arow0 = A + (size_t)(tm * 32 + ln) * CH;        // M-subtile 0, M=ln
    const bf16* arow1 = arow0 + (size_t)16 * CH;                // M-subtile 1
    const bf16* bcol0 = W + (size_t)(tn * 64 + ln) * CH;        // N-subtiles, N=ln
    const bf16* bcol1 = bcol0 + (size_t)16 * CH;
    const bf16* bcol2 = bcol0 + (size_t)32 * CH;
    const bf16* bcol3 = bcol0 + (size_t)48 * CH;

    v8f acc[2][4];
    #pragma unroll
    for (int i = 0; i < 2; ++i)
        #pragma unroll
        for (int j = 0; j < 4; ++j) acc[i][j] = (v8f){0.f,0.f,0.f,0.f,0.f,0.f,0.f,0.f};

    #pragma unroll
    for (int ks = 0; ks < 4; ++ks) {
        const int k0 = ks * 32;
        // A 16x32 frags: lane<16 -> K k0..k0+15; lane>=16 -> K k0+16..k0+31
        v16bf af[2];
        {
            v8bf a00 = *(const v8bf*)(arow0 + k0 + hi * 16);
            v8bf a01 = *(const v8bf*)(arow0 + k0 + hi * 16 + 8);
            v8bf a10 = *(const v8bf*)(arow1 + k0 + hi * 16);
            v8bf a11 = *(const v8bf*)(arow1 + k0 + hi * 16 + 8);
            #pragma unroll
            for (int i = 0; i < 8; ++i) {
                af[0][i] = a00[i]; af[0][8 + i] = a01[i];
                af[1][i] = a10[i]; af[1][8 + i] = a11[i];
            }
        }
        // B 32x16 frags: lane<16 -> K k0..k0+15; lane>=16 -> K k0+16..k0+31
        v16bf bfr[4];
        bfr[0] = *(const v16bf*)(bcol0 + k0 + hi * 16);
        bfr[1] = *(const v16bf*)(bcol1 + k0 + hi * 16);
        bfr[2] = *(const v16bf*)(bcol2 + k0 + hi * 16);
        bfr[3] = *(const v16bf*)(bcol3 + k0 + hi * 16);

        #pragma unroll
        for (int i = 0; i < 2; ++i)
            #pragma unroll
            for (int j = 0; j < 4; ++j)
                acc[i][j] = __builtin_amdgcn_wmma_f32_16x16x32_bf16(
                    false, af[i], false, bfr[j], (short)0, acc[i][j], false, false);
    }

    #pragma unroll
    for (int j = 0; j < 4; ++j) {
        const int n = tn * 64 + j * 16 + ln;
        const float bn = bias[n];
        #pragma unroll
        for (int i = 0; i < 2; ++i) {
            #pragma unroll
            for (int r = 0; r < 8; ++r) {
                int m = tm * 32 + i * 16 + r + hi * 8;
                float v = acc[i][j][r] + bn;
                if (MODE == 0) {
                    outF[(size_t)m * NCOLS + n] = v;
                } else if (MODE == 1) {
                    outB[(size_t)m * NCOLS + n] = f2bf(v);
                } else {
                    int b = m / (HH * WW);
                    int rem = m - b * (HH * WW);
                    int h = rem / WW, w = rem - h * WW;
                    int off = ((b * CH + n) * HH + h) * WW + w;
                    outF[off] = v + resid[off];
                }
            }
        }
    }
}

// One thread per (b, head, h, w); qkv [M, 3*CH] fp32; out [M, CH] bf16.
template<int K, int DIL, int RP>
__global__ void nat_attn(const float* __restrict__ qkv,
                         const float* __restrict__ rpb,
                         bf16* __restrict__ out)
{
    int t = blockIdx.x * blockDim.x + threadIdx.x;
    if (t >= BN * HEADS * HH * WW) return;
    int w = t % WW;
    int h = (t / WW) % HH;
    int g = (t / (WW * HH)) % HEADS;
    int b = t / (WW * HH * HEADS);

    int nh[K], ph[K], nw[K], pw[K];
    {
        const int ns = K / 2;
        int gg = h % DIL, idx = h / DIL;
        int lg = (HH - gg + DIL - 1) / DIL;
        int st = idx - ns; if (st < 0) st = 0; if (st > lg - K) st = lg - K;
        #pragma unroll
        for (int j = 0; j < K; ++j) { nh[j] = gg + DIL * (st + j); ph[j] = K - 1 + st - idx + j; }
        gg = w % DIL; idx = w / DIL;
        lg = (WW - gg + DIL - 1) / DIL;
        st = idx - ns; if (st < 0) st = 0; if (st > lg - K) st = lg - K;
        #pragma unroll
        for (int j = 0; j < K; ++j) { nw[j] = gg + DIL * (st + j); pw[j] = K - 1 + st - idx + j; }
    }

    const int m = (b * HH + h) * WW + w;
    const float4* qp = (const float4*)(qkv + (size_t)m * (3 * CH) + g * HD);
    float4 q[8];
    #pragma unroll
    for (int r = 0; r < 8; ++r) q[r] = qp[r];

    const float scale = 0.17677669529663687f; // 1/sqrt(HD)
    float sc[K * K];
    float mx = -1e30f;
    #pragma unroll
    for (int jh = 0; jh < K; ++jh) {
        #pragma unroll
        for (int jw = 0; jw < K; ++jw) {
            int km = (b * HH + nh[jh]) * WW + nw[jw];
            const float4* kp = (const float4*)(qkv + (size_t)km * (3 * CH) + CH + g * HD);
            float d = 0.f;
            #pragma unroll
            for (int r = 0; r < 8; ++r) {
                float4 kv = kp[r];
                d += q[r].x * kv.x + q[r].y * kv.y + q[r].z * kv.z + q[r].w * kv.w;
            }
            float s = scale * d + rpb[(g * RP + ph[jh]) * RP + pw[jw]];
            sc[jh * K + jw] = s;
            mx = fmaxf(mx, s);
        }
    }
    float sum = 0.f;
    #pragma unroll
    for (int j = 0; j < K * K; ++j) { float e = __expf(sc[j] - mx); sc[j] = e; sum += e; }
    const float inv = 1.f / sum;

    float4 acc[8];
    #pragma unroll
    for (int r = 0; r < 8; ++r) acc[r] = make_float4(0.f, 0.f, 0.f, 0.f);
    #pragma unroll
    for (int jh = 0; jh < K; ++jh) {
        #pragma unroll
        for (int jw = 0; jw < K; ++jw) {
            int km = (b * HH + nh[jh]) * WW + nw[jw];
            const float4* vp = (const float4*)(qkv + (size_t)km * (3 * CH) + 2 * CH + g * HD);
            float p = sc[jh * K + jw] * inv;
            #pragma unroll
            for (int r = 0; r < 8; ++r) {
                float4 vv = vp[r];
                acc[r].x += p * vv.x; acc[r].y += p * vv.y;
                acc[r].z += p * vv.z; acc[r].w += p * vv.w;
            }
        }
    }
    bf16* op = out + (size_t)m * CH + g * HD;
    #pragma unroll
    for (int r = 0; r < 8; ++r) {
        op[r * 4 + 0] = f2bf(acc[r].x);
        op[r * 4 + 1] = f2bf(acc[r].y);
        op[r * 4 + 2] = f2bf(acc[r].z);
        op[r * 4 + 3] = f2bf(acc[r].w);
    }
}

extern "C" void kernel_launch(void* const* d_in, const int* in_sizes, int n_in,
                              void* d_out, int out_size, void* d_ws, size_t ws_size,
                              hipStream_t stream) {
    const float* x       = (const float*)d_in[0];
    const float* qkv1_b  = (const float*)d_in[2];
    const float* proj1_b = (const float*)d_in[4];
    const float* rpb1    = (const float*)d_in[5];
    const float* qkv2_b  = (const float*)d_in[7];
    const float* proj2_b = (const float*)d_in[9];
    const float* rpb2    = (const float*)d_in[10];
    float* out = (float*)d_out;

    char* p = (char*)d_ws;
    auto carve = [&](size_t bytes) {
        char* r = p;
        p += (bytes + 255) & ~(size_t)255;
        return (void*)r;
    };
    bf16*  xb    = (bf16*) carve((size_t)MROWS * CH * 2);
    bf16*  wq1   = (bf16*) carve((size_t)3 * CH * CH * 2);
    bf16*  wp1   = (bf16*) carve((size_t)CH * CH * 2);
    bf16*  wq2   = (bf16*) carve((size_t)3 * CH * CH * 2);
    bf16*  wp2   = (bf16*) carve((size_t)CH * CH * 2);
    float* qkvb  = (float*)carve((size_t)MROWS * 3 * CH * 4);
    bf16*  attnb = (bf16*) carve((size_t)MROWS * CH * 2);
    bf16*  y1b   = (bf16*) carve((size_t)MROWS * CH * 2);

    // convert inputs/weights to bf16 once per call
    nat_cvt_x<<<(BN * CH * HH * WW + 255) / 256, 256, 0, stream>>>(x, xb);
    nat_cvt<<<(3 * CH * CH + 255) / 256, 256, 0, stream>>>((const float*)d_in[1], wq1, 3 * CH * CH);
    nat_cvt<<<(CH * CH + 255) / 256, 256, 0, stream>>>((const float*)d_in[3], wp1, CH * CH);
    nat_cvt<<<(3 * CH * CH + 255) / 256, 256, 0, stream>>>((const float*)d_in[6], wq2, 3 * CH * CH);
    nat_cvt<<<(CH * CH + 255) / 256, 256, 0, stream>>>((const float*)d_in[8], wp2, CH * CH);

    const int tilesQKV  = (MROWS / 32) * ((3 * CH) / 64);  // 576 * 6 = 3456
    const int tilesPROJ = (MROWS / 32) * (CH / 64);        // 576 * 2 = 1152
    const int attnThreads = BN * HEADS * HH * WW;

    // ---- layer 1: K=7, dil=1 ----
    nat_gemm<3 * CH, 0><<<(tilesQKV + 7) / 8, 256, 0, stream>>>(xb, wq1, qkv1_b, qkvb, nullptr, nullptr);
    nat_attn<7, 1, 13><<<(attnThreads + 127) / 128, 128, 0, stream>>>(qkvb, rpb1, attnb);
    nat_gemm<CH, 1><<<(tilesPROJ + 7) / 8, 256, 0, stream>>>(attnb, wp1, proj1_b, nullptr, y1b, nullptr);

    // ---- layer 2: K=5, dil=2, fused residual + transpose epilogue ----
    nat_gemm<3 * CH, 0><<<(tilesQKV + 7) / 8, 256, 0, stream>>>(y1b, wq2, qkv2_b, qkvb, nullptr, nullptr);
    nat_attn<5, 2, 9><<<(attnThreads + 127) / 128, 128, 0, stream>>>(qkvb, rpb2, attnb);
    nat_gemm<CH, 2><<<(tilesPROJ + 7) / 8, 256, 0, stream>>>(attnb, wp2, proj2_b, out, nullptr, x);
}
